// ExplicitSoftmaxAttention_78872779423754
// MI455X (gfx1250) — matrix operand order repaired and checked by reading, not compiled, not run on hardware
//
#include <hip/hip_runtime.h>
#include <hip/hip_bf16.h>

// ---------- vector types matching gfx1250 WMMA builtin signatures ----------
typedef __attribute__((ext_vector_type(16))) __bf16 bf16x16;
typedef __attribute__((ext_vector_type(8)))  __bf16 bf16x8;
typedef __attribute__((ext_vector_type(8)))  float  f32x8;
typedef int v4i __attribute__((vector_size(4 * sizeof(int))));  // matches builtin param

#define AS1 __attribute__((address_space(1)))
#define AS3 __attribute__((address_space(3)))

static __device__ __forceinline__ bf16x16 combine8(bf16x8 lo, bf16x8 hi) {
  return __builtin_shufflevector(lo, hi, 0,1,2,3,4,5,6,7,8,9,10,11,12,13,14,15);
}

static __device__ __forceinline__ f32x8 wmma_bf16(bf16x16 a, bf16x16 b, f32x8 c) {
  // D(16x16,f32) = A(16x32,bf16) x B(32x16,bf16) + C
  return __builtin_amdgcn_wmma_f32_16x16x32_bf16(
      /*neg_a=*/false, a, /*neg_b=*/false, b,
      /*c_mod=*/(short)0, c, /*reuse_a=*/false, /*reuse_b=*/false);
}

// 16-byte async copy global->LDS (bf16 data, no conversion). Tracked by ASYNCcnt.
static __device__ __forceinline__ void async_copy_b128(const void* g, void* l) {
  __builtin_amdgcn_global_load_async_to_lds_b128(
      (AS1 v4i*)g, (AS3 v4i*)l, 0, 0);
}
static __device__ __forceinline__ void wait_asynccnt0() {
  asm volatile("s_wait_asynccnt 0" ::: "memory");
}

static __device__ __forceinline__ float ldf(const float* p)  { return *p; }
static __device__ __forceinline__ float ldf(const __bf16* p) { return (float)*p; }
static __device__ __forceinline__ void  stf(float* p, float v)  { *p = v; }
static __device__ __forceinline__ void  stf(__bf16* p, float v) { *p = (__bf16)v; }

// =====================================================================
// Generic GEMM:  C[M,N] = A[M,K] * B[K,N] + bias[N]
// A: f32 or bf16 (converted to bf16 in LDS), B: f32 (converted), C: f32 or bf16.
// Block: 256 threads (8 waves), 128x128 tile, BK=32.
// Wave grid 2(M) x 4(N): each wave owns 64x32 -> 4x2 WMMA accumulators.
// =====================================================================
#define LDSK 40   // 32 + 8 pad (bf16 elems); row stride = 80B (16B aligned)

template<typename AT, typename OT>
__global__ __launch_bounds__(256) void gemm_bias_kernel(
    const AT* __restrict__ A, const float* __restrict__ B,
    const float* __restrict__ bias, OT* __restrict__ C,
    int M, int N, int K)
{
  __shared__ alignas(16) __bf16 As[128 * LDSK];   // [m][k]
  __shared__ alignas(16) __bf16 Bt[128 * LDSK];   // [n][k] (transposed)

  const int tid   = threadIdx.x;
  const int lane  = tid & 31;
  const int wave  = tid >> 5;
  const int l16   = lane & 15;
  const int lhalf = lane >> 4;          // 0 or 1
  const int wm    = wave & 1;           // 64-row slab
  const int wn    = wave >> 1;          // 32-col slab
  const int mBase = blockIdx.y * 128;
  const int nBase = blockIdx.x * 128;

  f32x8 acc[4][2];
#pragma unroll
  for (int mi = 0; mi < 4; ++mi)
#pragma unroll
    for (int ni = 0; ni < 2; ++ni) acc[mi][ni] = (f32x8){};

  for (int kb = 0; kb < K; kb += 32) {
    __syncthreads();
    // ---- stage A (128x32) and B^T (128x32) into LDS as bf16, coalesced ----
#pragma unroll
    for (int t = 0; t < 16; ++t) {
      int li = t * 256 + tid;                // 0..4095
      int am = li >> 5, ak = li & 31;        // A: row-major
      As[am * LDSK + ak] = (__bf16)ldf(&A[(size_t)(mBase + am) * K + kb + ak]);
      int bk = li >> 7, bn = li & 127;       // B: read row-major, write transposed
      Bt[bn * LDSK + bk] = (__bf16)B[(size_t)(kb + bk) * N + nBase + bn];
    }
    if (kb + 32 < K) {  // next-tile prefetch -> global_prefetch_b8
      __builtin_prefetch(&A[(size_t)(mBase + (tid >> 1)) * K + kb + 32], 0, 1);
      __builtin_prefetch(&B[(size_t)(kb + 32 + (tid >> 7)) * N + nBase + (tid & 127)], 0, 1);
    }
    __syncthreads();

    // ---- build fragments per ISA VGPR layouts ----
    bf16x16 af[4], bfr[2];
#pragma unroll
    for (int mi = 0; mi < 4; ++mi) {
      int row = wm * 64 + mi * 16 + l16;
      int ka  = lhalf * 8;                    // lanes0-15: K 0..7/16..23 ; lanes16-31: 8..15/24..31
      af[mi] = combine8(*(const bf16x8*)&As[row * LDSK + ka],
                        *(const bf16x8*)&As[row * LDSK + ka + 16]);
    }
#pragma unroll
    for (int ni = 0; ni < 2; ++ni) {
      int col = wn * 32 + ni * 16 + l16;
      int kbb = lhalf * 8;                    // lanes0-15: K 0..7/16..23 ; lanes16-31: 8..15/24..31
      bfr[ni] = combine8(*(const bf16x8*)&Bt[col * LDSK + kbb],
                         *(const bf16x8*)&Bt[col * LDSK + kbb + 16]);
    }
#pragma unroll
    for (int mi = 0; mi < 4; ++mi)
#pragma unroll
      for (int ni = 0; ni < 2; ++ni)
        acc[mi][ni] = wmma_bf16(af[mi], bfr[ni], acc[mi][ni]);
  }

  // ---- epilogue: C-layout (VGPR r -> row r or r+8 by lane half, lane -> col) ----
#pragma unroll
  for (int mi = 0; mi < 4; ++mi)
#pragma unroll
    for (int ni = 0; ni < 2; ++ni)
#pragma unroll
      for (int r = 0; r < 8; ++r) {
        int row = mBase + wm * 64 + mi * 16 + r + lhalf * 8;
        int col = nBase + wn * 32 + ni * 16 + l16;
        stf(&C[(size_t)row * N + col], acc[mi][ni][r] + bias[col]);
      }
}

// =====================================================================
// Flash attention over qkv (bf16, [8192][3072], col = c*1024 + h*64 + d).
// Block = one (b,h) and 128 query rows; 8 waves x 16 rows.
// K tile staged via GLOBAL_LOAD_ASYNC_TO_LDS_B128 (pure bf16 copy, ASYNCcnt);
// V tile transposed via scalar DS stores; online softmax; output bf16.
// Softmax scale 1/sqrt(64)=0.125 is a power of two -> folded exactly into Q.
// =====================================================================
#define SEQ   2048
#define QKVW  3072
#define HD    64
#define KV_S  72      // LDS stride: 144B, 16B aligned
#define NEG_BIG (-1.0e30f)

__global__ __launch_bounds__(256) void flash_attn_kernel(
    const __bf16* __restrict__ qkv, __bf16* __restrict__ attn_out)
{
  __shared__ alignas(16) __bf16 Kl[64 * KV_S];      // [key][d]
  __shared__ alignas(16) __bf16 Vt[64 * KV_S];      // [d][key] (transposed)
  __shared__ alignas(16) __bf16 Pl[8][16 * KV_S];   // per-wave P scratch [q][key]

  const int tid   = threadIdx.x;
  const int lane  = tid & 31;
  const int wave  = tid >> 5;
  const int l16   = lane & 15;
  const int lhalf = lane >> 4;
  const int rowOff = lhalf * 8;

  const int bh = blockIdx.x >> 4;      // 0..63
  const int qt = blockIdx.x & 15;      // query tile
  const int b  = bh >> 4;
  const int h  = bh & 15;

  // ---- load this wave's 16x64 Q tile as two A-fragments, pre-scaled ----
  const int    qRow  = qt * 128 + wave * 16 + l16;
  const size_t qBase = (size_t)(b * SEQ + qRow) * QKVW + h * HD;   // c=0 (Q)
  const int    ka    = lhalf * 8;
  bf16x16 qf[2];
#pragma unroll
  for (int c = 0; c < 2; ++c) {
    bf16x16 q = combine8(*(const bf16x8*)(qkv + qBase + c * 32 + ka),
                         *(const bf16x8*)(qkv + qBase + c * 32 + ka + 16));
#pragma unroll
    for (int e = 0; e < 16; ++e) q[e] = (__bf16)((float)q[e] * 0.125f);  // exact
    qf[c] = q;
  }

  f32x8 o[4];
#pragma unroll
  for (int d16 = 0; d16 < 4; ++d16) o[d16] = (f32x8){};
  float m[8], l[8];
#pragma unroll
  for (int r = 0; r < 8; ++r) { m[r] = NEG_BIG; l[r] = 0.0f; }

  for (int j0 = 0; j0 < SEQ; j0 += 64) {
    __syncthreads();
    // ---- K tile [key][d]: async bf16 copy (2x B128 per thread, 64x64 elems) ----
#pragma unroll
    for (int t = 0; t < 2; ++t) {
      int g  = t * 256 + tid;            // 512 groups of 8 bf16 (16B)
      int kk = g >> 3, d8 = (g & 7) * 8;
      async_copy_b128(qkv + (size_t)(b * SEQ + j0 + kk) * QKVW + 1024 + h * HD + d8,
                      &Kl[kk * KV_S + d8]);
    }
    // ---- V tile transposed [d][key]: scalar DS stores ----
#pragma unroll
    for (int t = 0; t < 16; ++t) {
      int li = t * 256 + tid;
      int kk = li >> 6, dd = li & 63;
      Vt[dd * KV_S + kk] =
          qkv[(size_t)(b * SEQ + j0 + kk) * QKVW + 2048 + h * HD + dd];
    }
    wait_asynccnt0();     // ensure this wave's async LDS writes landed
    __syncthreads();

    // ---- S = Q * K^T  (4 tiles of 16q x 16k, two chained WMMAs each) ----
    f32x8 s[4];
#pragma unroll
    for (int kt = 0; kt < 4; ++kt) {
      int key = kt * 16 + l16;
      f32x8 a = (f32x8){};
#pragma unroll
      for (int c = 0; c < 2; ++c) {
        int db = c * 32 + lhalf * 8;
        bf16x16 kf = combine8(*(const bf16x8*)&Kl[key * KV_S + db],
                              *(const bf16x8*)&Kl[key * KV_S + db + 16]);
        a = wmma_bf16(qf[c], kf, a);
      }
      s[kt] = a;
    }

    // ---- online softmax (row reductions across 16-lane half-groups) ----
    float alpha[8];
#pragma unroll
    for (int r = 0; r < 8; ++r) {
      float v = NEG_BIG;
#pragma unroll
      for (int kt = 0; kt < 4; ++kt) v = fmaxf(v, s[kt][r]);
#pragma unroll
      for (int off = 1; off < 16; off <<= 1) v = fmaxf(v, __shfl_xor(v, off, 32));
      float mn = fmaxf(m[r], v);
      alpha[r] = __expf(m[r] - mn);
      m[r] = mn;
    }
#pragma unroll
    for (int r = 0; r < 8; ++r) {
      float rs = 0.0f;
#pragma unroll
      for (int kt = 0; kt < 4; ++kt) {
        float p = __expf(s[kt][r] - m[r]);
        rs += p;
        Pl[wave][(r + rowOff) * KV_S + kt * 16 + l16] = (__bf16)p;
      }
#pragma unroll
      for (int off = 1; off < 16; off <<= 1) rs += __shfl_xor(rs, off, 32);
      l[r] = l[r] * alpha[r] + rs;
    }
#pragma unroll
    for (int d16 = 0; d16 < 4; ++d16)
#pragma unroll
      for (int r = 0; r < 8; ++r) o[d16][r] *= alpha[r];

    // ---- O += P * V (P via per-wave LDS round-trip; LDS is in-order per wave) ----
    bf16x16 pf[2];
#pragma unroll
    for (int c = 0; c < 2; ++c) {
      int kbp = c * 32 + ka;
      pf[c] = combine8(*(const bf16x8*)&Pl[wave][l16 * KV_S + kbp],
                       *(const bf16x8*)&Pl[wave][l16 * KV_S + kbp + 16]);
    }
#pragma unroll
    for (int d16 = 0; d16 < 4; ++d16) {
      int dcol = d16 * 16 + l16;
#pragma unroll
      for (int c = 0; c < 2; ++c) {
        int kbv = c * 32 + lhalf * 8;
        bf16x16 vf = combine8(*(const bf16x8*)&Vt[dcol * KV_S + kbv],
                              *(const bf16x8*)&Vt[dcol * KV_S + kbv + 16]);
        o[d16] = wmma_bf16(pf[c], vf, o[d16]);
      }
    }
  }

  // ---- normalize and emit bf16 [8192][1024] (col = h*64 + d) ----
#pragma unroll
  for (int d16 = 0; d16 < 4; ++d16)
#pragma unroll
    for (int r = 0; r < 8; ++r) {
      int q = qt * 128 + wave * 16 + r + rowOff;
      int d = d16 * 16 + l16;
      float val = o[d16][r] / l[r];
      attn_out[(size_t)(b * SEQ + q) * 1024 + h * HD + d] = (__bf16)val;
    }
}

// =====================================================================
extern "C" void kernel_launch(void* const* d_in, const int* in_sizes, int n_in,
                              void* d_out, int out_size, void* d_ws, size_t ws_size,
                              hipStream_t stream) {
  const float* x     = (const float*)d_in[0];   // [4,2048,1024]
  const float* W_qkv = (const float*)d_in[1];   // [1024,3072]
  const float* b_qkv = (const float*)d_in[2];   // [3072]
  const float* W_out = (const float*)d_in[3];   // [1024,1024]
  const float* b_out = (const float*)d_in[4];   // [1024]
  float* out = (float*)d_out;                   // [4,2048,1024] f32

  __bf16* qkv_ws  = (__bf16*)d_ws;                       // 8192*3072 bf16 (48MB)
  __bf16* attn_ws = qkv_ws + (size_t)8192 * 3072;        // 8192*1024 bf16 (16MB)

  dim3 blk(256);
  // 1) QKV projection: [8192,1024] x [1024,3072] -> bf16 ws
  gemm_bias_kernel<float, __bf16><<<dim3(3072 / 128, 8192 / 128), blk, 0, stream>>>(
      x, W_qkv, b_qkv, qkv_ws, 8192, 3072, 1024);
  // 2) Flash attention per (b,h,qtile)
  flash_attn_kernel<<<dim3(64 * 16), blk, 0, stream>>>(qkv_ws, attn_ws);
  // 3) Output projection: [8192,1024] x [1024,1024] -> f32 out
  gemm_bias_kernel<__bf16, float><<<dim3(1024 / 128, 8192 / 128), blk, 0, stream>>>(
      attn_ws, W_out, b_out, out, 8192, 1024, 1024);
}
